// GETLayer_25237227831479
// MI455X (gfx1250) — matrix operand order repaired and checked by reading, not compiled, not run on hardware
//
#include <hip/hip_runtime.h>

typedef unsigned short u16;
typedef unsigned int   u32;

typedef __attribute__((ext_vector_type(16))) __bf16 v16bf;
typedef __attribute__((ext_vector_type(8)))  float  v8f;
typedef __attribute__((ext_vector_type(2)))  float  v2f;
typedef __attribute__((ext_vector_type(4)))  int    v4i;

#define NROW   32768
#define DIM    256
#define NC     1792      // packed output columns: [Q2 |K2 |Q3(512)|K3(512)|Qm]
#define SCALE  0.0625f   // 1/sqrt(256)

// ---------------- CDNA5 async global->LDS copy (ASYNCcnt path) ----------------
#if defined(__has_builtin)
#if __has_builtin(__builtin_amdgcn_global_load_async_to_lds_b128) && \
    __has_builtin(__builtin_amdgcn_s_wait_asynccnt)
#define USE_ASYNC_LDS 1
#endif
#endif

#ifdef USE_ASYNC_LDS
#define GLOBAL_AS __attribute__((address_space(1)))
#define LDS_AS    __attribute__((address_space(3)))
__device__ __forceinline__ void cp16(void* lds_dst, const void* gsrc){
  // prototype (from compiler diagnostic): pointers to int __vector(4)
  __builtin_amdgcn_global_load_async_to_lds_b128(
      (GLOBAL_AS v4i*)gsrc, (LDS_AS v4i*)lds_dst, 0, 0);
}
__device__ __forceinline__ void cp_wait(){ __builtin_amdgcn_s_wait_asynccnt(0); }
#else
__device__ __forceinline__ void cp16(void* lds_dst, const void* gsrc){
  *(uint4*)lds_dst = *(const uint4*)gsrc;
}
__device__ __forceinline__ void cp_wait(){}
#endif

// ---------------- helpers ----------------
__device__ __forceinline__ float bflo(u32 w){ return __uint_as_float(w << 16); }
__device__ __forceinline__ float bfhi(u32 w){ return __uint_as_float(w & 0xFFFF0000u); }
__device__ __forceinline__ u16 f2bf(float f){
  u32 u = __float_as_uint(f);
  u32 r = u + 0x7FFFu + ((u >> 16) & 1u);
  return (u16)(r >> 16);
}
// order-preserving float <-> uint (so atomicMax on u32 with 0-init works)
__device__ __forceinline__ u32 f2ord(float f){
  u32 u = __float_as_uint(f);
  return (u & 0x80000000u) ? ~u : (u | 0x80000000u);
}
__device__ __forceinline__ float ord2f(u32 o){
  u32 u = (o & 0x80000000u) ? (o & 0x7FFFFFFFu) : ~o;
  return __uint_as_float(u);
}
__device__ __forceinline__ void bf8(const u16* p, float* f){
  uint4 v = *(const uint4*)p;
  f[0]=bflo(v.x); f[1]=bfhi(v.x); f[2]=bflo(v.y); f[3]=bfhi(v.y);
  f[4]=bflo(v.z); f[5]=bfhi(v.z); f[6]=bflo(v.w); f[7]=bfhi(v.w);
}
__device__ __forceinline__ float waveSum(float v){
#pragma unroll
  for (int o = 16; o > 0; o >>= 1) v += __shfl_xor(v, o, 32);
  return v;
}
__device__ __forceinline__ float waveMax(float v){
#pragma unroll
  for (int o = 16; o > 0; o >>= 1) v = fmaxf(v, __shfl_xor(v, o, 32));
  return v;
}
__device__ __forceinline__ float blockSum256(float v, float* sbuf){
  const int tid = threadIdx.x;
  v = waveSum(v);
  if ((tid & 31) == 0) sbuf[tid >> 5] = v;
  __syncthreads();
  if (tid == 0){
    float t = 0.f;
#pragma unroll
    for (int i = 0; i < 8; ++i) t += sbuf[i];
    sbuf[8] = t;
  }
  __syncthreads();
  float r = sbuf[8];
  __syncthreads();
  return r;
}

// ---------------- scalar params ----------------
// P: [0]=lam2 [1]=lam3 [2]=lamm [3]=b2 [4]=b3 [5]=bm [6]=step*DAMP
__global__ void k_params(const float* step, const float* l2, const float* l3,
                         const float* lm, const float* pb2, const float* pb3,
                         const float* pbm, float* P){
  auto sp = [](float x){ return (x > 20.f) ? x : log1pf(expf(x)); };
  P[0] = sp(*l2); P[1] = sp(*l3); P[2] = sp(*lm);
  P[3] = fminf(sp(*pb2), 5.f);
  P[4] = fminf(sp(*pb3), 5.f);
  P[5] = fminf(sp(*pbm), 5.f);
  P[6] = (*step) * 0.9999f;
}

// ---------------- weight prep: fused W (bf16) + W^T (fp32) ----------------
__global__ void k_prep_w(const float* __restrict__ WQ2, const float* __restrict__ WK2,
                         const float* __restrict__ WQ3, const float* __restrict__ WK3,
                         const float* __restrict__ WQm,
                         u16* __restrict__ Wbf, float* __restrict__ WT){
  int idx = blockIdx.x * 256 + threadIdx.x;
  if (idx >= DIM * NC) return;
  int i = idx / NC, j = idx % NC;
  float w;
  if (j < 256)       w = WQ2[i * 256 + j];
  else if (j < 512)  w = WK2[i * 256 + (j - 256)];
  else if (j < 1024) w = WQ3[i * 512 + (j - 512)];
  else if (j < 1536) w = WK3[i * 512 + (j - 1024)];
  else               w = WQm[i * 256 + (j - 1536)];
  Wbf[idx] = f2bf(w);
  WT[(size_t)j * DIM + i] = w;
}

// Km = B_mem @ W_Km   [32 x 256], tiny
__global__ void k_prep_km(const float* __restrict__ Bmem, const float* __restrict__ WKm,
                          float* __restrict__ Km){
  int idx = blockIdx.x * 256 + threadIdx.x;     // 8192 total
  int k = idx >> 8, d = idx & 255;
  float acc = 0.f;
  for (int i = 0; i < 256; ++i) acc += Bmem[k * 256 + i] * WKm[i * 256 + d];
  Km[idx] = acc;
}

// ---------------- LayerNorm -> bf16 G ----------------
__global__ void k_ln(const float* __restrict__ X, const float* __restrict__ gamma,
                     const float* __restrict__ beta, u16* __restrict__ Gbf,
                     float* __restrict__ murstd){
  __shared__ float sbuf[16];
  const int n = blockIdx.x, d = threadIdx.x;
  float x = X[(size_t)n * DIM + d];
  float mu = blockSum256(x, sbuf) * (1.f / DIM);
  float xc = x - mu;
  float var = blockSum256(xc * xc, sbuf) * (1.f / DIM);
  float rstd = rsqrtf(var + 1e-5f);
  Gbf[(size_t)n * DIM + d] = f2bf(xc * rstd * gamma[d] + beta[d]);
  if (d == 0){ murstd[2 * n] = mu; murstd[2 * n + 1] = rstd; }
}

// ---------------- forward GEMM: Y(bf16) = Gbf[M,256] @ Wbf[256,1792] ----------------
// 8 waves/block; block tile 128(M) x 64(N) = 4 wmma sub-tiles; K step 32
__global__ void k_gemm_bf16(const u16* __restrict__ A, const u16* __restrict__ B,
                            u16* __restrict__ C){
  __shared__ __align__(16) u16 sA[128 * 32];   // 8 KB
  __shared__ __align__(16) u16 sB[32 * 64];    // 4 KB
  const int tid  = threadIdx.x;
  const int wave = tid >> 5, lane = tid & 31;
  const int mBlk = blockIdx.x * 128;
  const int nBlk = blockIdx.y * 64;
  v8f acc[4] = {};
  for (int k0 = 0; k0 < DIM; k0 += 32){
    { // A tile 128x32 halves: 2x 16B chunks per thread
      int row  = tid >> 1;
      int hoff = (tid & 1) * 16;
      const u16* src = A + (size_t)(mBlk + row) * DIM + k0 + hoff;
      u16* dst = sA + row * 32 + hoff;
      cp16(dst, src);
      cp16(dst + 8, src + 8);
    }
    { // B tile 32x64 halves: 1x 16B chunk per thread
      int row  = tid >> 3;
      int hoff = (tid & 7) * 8;
      cp16(sB + row * 64 + hoff, B + (size_t)(k0 + row) * NC + nBlk + hoff);
    }
    cp_wait();
    __syncthreads();
    union { v16bf v; u16 u[16]; } a, b;
    const int m  = lane & 15;
    const int kA = (lane < 16) ? 0 : 8;          // doc 16-bit A 16x32 layout
    const u16* arow = sA + (wave * 16 + m) * 32;
#pragma unroll
    for (int h = 0; h < 8; ++h){
      a.u[h]     = arow[kA + h];
      a.u[h + 8] = arow[kA + 16 + h];
    }
#pragma unroll
    for (int nt = 0; nt < 4; ++nt){
      const u16* brow = sB + nt * 16 + m;   // lane%16 = N column, elements = K (same order as A)
#pragma unroll
      for (int h = 0; h < 8; ++h){
        b.u[h]     = brow[(kA + h) * 64];
        b.u[h + 8] = brow[(kA + 16 + h) * 64];
      }
      acc[nt] = __builtin_amdgcn_wmma_f32_16x16x32_bf16(false, a.v, false, b.v,
                                                        (short)0, acc[nt], false, false);
    }
    __syncthreads();
  }
  const int n = lane & 15;
  const int mBase = mBlk + wave * 16 + ((lane < 16) ? 0 : 8);
#pragma unroll
  for (int nt = 0; nt < 4; ++nt)
#pragma unroll
    for (int r = 0; r < 8; ++r)
      C[(size_t)(mBase + r) * NC + nBlk + nt * 16 + n] = f2bf(acc[nt][r]);
}

// ---------------- backward GEMM: dG = dY[M,1792] @ WT[1792,256], fp32 WMMA ----------------
// block tile 128(M) x 128(N) = 8 sub-tiles -> dY streamed from HBM only 2x
__global__ void k_gemm_f32(const float* __restrict__ A, const float* __restrict__ B,
                           float* __restrict__ C){
  __shared__ __align__(16) float sA[128 * 32];   // 16 KB
  __shared__ __align__(16) float sB[32 * 128];   // 16 KB
  const int tid  = threadIdx.x;
  const int wave = tid >> 5, lane = tid & 31;
  const int mBlk = blockIdx.x * 128;
  const int nBlk = blockIdx.y * 128;
  v8f acc[8] = {};
  for (int k0 = 0; k0 < NC; k0 += 32){
    { // A tile 128x32 f32: 4x 16B chunks per thread
      int row  = tid >> 1;
      int coff = (tid & 1) * 16;
      const float* src = A + (size_t)(mBlk + row) * NC + k0 + coff;
      float* dst = sA + row * 32 + coff;
      cp16(dst, src); cp16(dst + 4, src + 4); cp16(dst + 8, src + 8); cp16(dst + 12, src + 12);
    }
    { // B tile 32x128 f32: 4x 16B chunks per thread
      int row  = tid >> 3;
      int coff = (tid & 7) * 16;
      const float* src = B + (size_t)(k0 + row) * DIM + nBlk + coff;
      float* dst = sB + row * 128 + coff;
      cp16(dst, src); cp16(dst + 4, src + 4); cp16(dst + 8, src + 8); cp16(dst + 12, src + 12);
    }
    cp_wait();
    __syncthreads();
    const int m = lane & 15;
    const float* arow = sA + (wave * 16 + m) * 32;
#pragma unroll
    for (int kk = 0; kk < 32; kk += 4){
      v2f a;
      a.x = arow[kk + ((lane < 16) ? 0 : 2)];   // doc 32-bit A 16x4 layout
      a.y = arow[kk + ((lane < 16) ? 1 : 3)];
      const int kb0 = (kk + ((lane < 16) ? 0 : 1)) * 128 + m;
      const int kb1 = (kk + ((lane < 16) ? 2 : 3)) * 128 + m;
#pragma unroll
      for (int nt = 0; nt < 8; ++nt){
        v2f b;
        b.x = sB[kb0 + nt * 16];
        b.y = sB[kb1 + nt * 16];
        acc[nt] = __builtin_amdgcn_wmma_f32_16x16x4_f32(false, a, false, b,
                                                        (short)0, acc[nt], false, false);
      }
    }
    __syncthreads();
  }
  const int n = lane & 15;
  const int mBase = mBlk + wave * 16 + ((lane < 16) ? 0 : 8);
#pragma unroll
  for (int nt = 0; nt < 8; ++nt)
#pragma unroll
    for (int r = 0; r < 8; ++r)
      C[(size_t)(mBase + r) * DIM + nBlk + nt * 16 + n] = acc[nt][r];
}

// ---------------- order-2 edges ----------------
__global__ void k_edge_dot(const u16* __restrict__ Ybf, const int* __restrict__ c2,
                           const int* __restrict__ u2, const float* __restrict__ P,
                           float* __restrict__ s2, u32* __restrict__ m2u, int ne){
  const int wave = threadIdx.x >> 5, lane = threadIdx.x & 31;
  const long long e = (long long)blockIdx.x * 8 + wave;
  if (e >= ne) return;
  const int c = c2[e], u = u2[e];
  float q[8], k[8];
  bf8(Ybf + (size_t)c * NC + lane * 8, q);          // Q2 cols [0,256)
  bf8(Ybf + (size_t)u * NC + DIM + lane * 8, k);    // K2 cols [256,512)
  float dot = 0.f;
#pragma unroll
  for (int i = 0; i < 8; ++i) dot += q[i] * k[i];
  dot = waveSum(dot);
  if (lane == 0){
    float s = P[3] * SCALE * dot;
    s2[e] = s;
    atomicMax(&m2u[c], f2ord(s));
  }
}

__global__ void k_seg_sum(const float* __restrict__ s, const int* __restrict__ seg,
                          const u32* __restrict__ mu, float* __restrict__ ebuf, int n){
  int e = blockIdx.x * 256 + threadIdx.x;
  if (e >= n) return;
  int c = seg[e];
  atomicAdd(&ebuf[c], expf(s[e] - ord2f(mu[c])));
}

__global__ void k_seg_lse(const u32* __restrict__ mu, const float* __restrict__ ebuf,
                          float* __restrict__ lse, float* __restrict__ eslot){
  __shared__ float sbuf[16];
  int n = blockIdx.x * 256 + threadIdx.x;
  float ev = ebuf[n];
  float v = (ev > 0.f) ? (ord2f(mu[n]) + logf(fmaxf(ev, 1e-30f))) : 0.f;
  lse[n] = v;
  float t = blockSum256(v, sbuf);
  if (threadIdx.x == 0) atomicAdd(eslot, t);
}

__global__ void k_edge_bwd(const u16* __restrict__ Ybf, const int* __restrict__ c2,
                           const int* __restrict__ u2, const float* __restrict__ s2,
                           const float* __restrict__ lse2, const float* __restrict__ P,
                           float* __restrict__ dY, int ne){
  const int wave = threadIdx.x >> 5, lane = threadIdx.x & 31;
  const long long e = (long long)blockIdx.x * 8 + wave;
  if (e >= ne) return;
  const int c = c2[e], u = u2[e];
  const float coef = -P[0] * SCALE * expf(s2[e] - lse2[c]);
  float q[8], k[8];
  bf8(Ybf + (size_t)c * NC + lane * 8, q);
  bf8(Ybf + (size_t)u * NC + DIM + lane * 8, k);
  float* dQ = dY + (size_t)c * NC + lane * 8;
  float* dK = dY + (size_t)u * NC + DIM + lane * 8;
#pragma unroll
  for (int i = 0; i < 8; ++i){
    atomicAdd(dQ + i, coef * k[i]);
    atomicAdd(dK + i, coef * q[i]);
  }
}

// ---------------- order-3 triangles ----------------
__global__ void k_tri_dot(const u16* __restrict__ Ybf, const int* __restrict__ c3,
                          const int* __restrict__ u3, const int* __restrict__ v3,
                          const int* __restrict__ tt, const float* __restrict__ Ttau,
                          const float* __restrict__ P, float* __restrict__ s3,
                          u32* __restrict__ m3u, int nt){
  const int wave = threadIdx.x >> 5, lane = threadIdx.x & 31;
  const long long e = (long long)blockIdx.x * 8 + wave;
  if (e >= nt) return;
  const int c = c3[e], u = u3[e], v = v3[e], t = tt[e];
  const int j0 = lane * 16;
  float q[16], ku[16], kv[16];
  bf8(Ybf + (size_t)c * NC + 512 + j0, q);   bf8(Ybf + (size_t)c * NC + 512 + j0 + 8, q + 8);
  bf8(Ybf + (size_t)u * NC + 1024 + j0, ku); bf8(Ybf + (size_t)u * NC + 1024 + j0 + 8, ku + 8);
  bf8(Ybf + (size_t)v * NC + 1024 + j0, kv); bf8(Ybf + (size_t)v * NC + 1024 + j0 + 8, kv + 8);
  const float* T = Ttau + (size_t)t * 512 + j0;
  float dot = 0.f;
#pragma unroll
  for (int i = 0; i < 16; ++i) dot += q[i] * ku[i] * kv[i] * T[i];
  dot = waveSum(dot);
  if (lane == 0){
    float s = P[4] * SCALE * dot;
    s3[e] = s;
    atomicMax(&m3u[c], f2ord(s));
  }
}

__global__ void k_tri_bwd(const u16* __restrict__ Ybf, const int* __restrict__ c3,
                          const int* __restrict__ u3, const int* __restrict__ v3,
                          const int* __restrict__ tt, const float* __restrict__ Ttau,
                          const float* __restrict__ s3, const float* __restrict__ lse3,
                          const float* __restrict__ P, float* __restrict__ dY, int nt){
  const int wave = threadIdx.x >> 5, lane = threadIdx.x & 31;
  const long long e = (long long)blockIdx.x * 8 + wave;
  if (e >= nt) return;
  const int c = c3[e], u = u3[e], v = v3[e], t = tt[e];
  const int j0 = lane * 16;
  float q[16], ku[16], kv[16];
  bf8(Ybf + (size_t)c * NC + 512 + j0, q);   bf8(Ybf + (size_t)c * NC + 512 + j0 + 8, q + 8);
  bf8(Ybf + (size_t)u * NC + 1024 + j0, ku); bf8(Ybf + (size_t)u * NC + 1024 + j0 + 8, ku + 8);
  bf8(Ybf + (size_t)v * NC + 1024 + j0, kv); bf8(Ybf + (size_t)v * NC + 1024 + j0 + 8, kv + 8);
  const float* T = Ttau + (size_t)t * 512 + j0;
  const float coef = -P[1] * SCALE * expf(s3[e] - lse3[c]);
  float* dQ  = dY + (size_t)c * NC + 512 + j0;
  float* dKu = dY + (size_t)u * NC + 1024 + j0;
  float* dKv = dY + (size_t)v * NC + 1024 + j0;
#pragma unroll
  for (int i = 0; i < 16; ++i){
    float Ti = T[i];
    atomicAdd(dQ + i,  coef * ku[i] * kv[i] * Ti);
    atomicAdd(dKu + i, coef * q[i]  * kv[i] * Ti);
    atomicAdd(dKv + i, coef * q[i]  * ku[i] * Ti);
  }
}

// ---------------- memory-slot term: fwd + bwd fused (K=32) ----------------
__global__ void k_mem(const u16* __restrict__ Ybf, const float* __restrict__ Km,
                      const float* __restrict__ P, float* __restrict__ dY,
                      float* __restrict__ eslot){
  __shared__ float sKm[32][257];   // +1 pad -> conflict-free row reads
  __shared__ float sQ[8][256];
  __shared__ float sP[8][32];
  const int tid = threadIdx.x, wave = tid >> 5, lane = tid & 31;
  for (int i = tid; i < 32 * 256; i += 256) sKm[i >> 8][i & 255] = Km[i];
  const int n = blockIdx.x * 8 + wave;
  bf8(Ybf + (size_t)n * NC + 1536 + lane * 8, &sQ[wave][lane * 8]);  // Qm cols
  __syncthreads();
  float s = 0.f;
  for (int d = 0; d < 256; ++d) s += sQ[wave][d] * sKm[lane][d];
  s *= P[5] * SCALE;
  float mx = waveMax(s);
  float ex = expf(s - mx);
  float se = waveSum(ex);
  sP[wave][lane] = ex / se;
  if (lane == 0) atomicAdd(eslot, mx + logf(se));
  __syncthreads();
  const float coef = -P[2] * SCALE;
  float* drow = dY + (size_t)n * NC + 1536 + lane * 8;
#pragma unroll
  for (int i = 0; i < 8; ++i){
    int d = lane * 8 + i;
    float acc = 0.f;
    for (int k = 0; k < 32; ++k) acc += sP[wave][k] * sKm[k][d];
    drow[i] = coef * acc;   // sole writer of this region
  }
}

// ---------------- LN backward + grad clip + update + state clip ----------------
__global__ void k_update(const float* __restrict__ X, const float* __restrict__ gamma,
                         const float* __restrict__ murstd, const float* __restrict__ dG,
                         const float* __restrict__ P, float* __restrict__ out){
  __shared__ float sbuf[16];
  const int n = blockIdx.x, d = threadIdx.x;
  const float mu = murstd[2 * n], rstd = murstd[2 * n + 1];
  const float x = X[(size_t)n * DIM + d];
  const float xhat = (x - mu) * rstd;
  const float dxh = dG[(size_t)n * DIM + d] * gamma[d];
  const float m1 = blockSum256(dxh, sbuf) * (1.f / DIM);
  const float m2 = blockSum256(dxh * xhat, sbuf) * (1.f / DIM);
  float g = rstd * (dxh - m1 - xhat * m2);
  const float gn = fmaxf(sqrtf(blockSum256(g * g, sbuf)), 1e-6f);
  g *= fminf(1.f / gn, 1.f);                       // GRAD_CLIP = 1
  float xn = x - P[6] * g;                          // step*DAMP
  const float sn = fmaxf(sqrtf(blockSum256(xn * xn, sbuf)), 1e-6f);
  out[(size_t)n * DIM + d] = xn * fminf(10.f / sn, 1.f);  // STATE_CLIP = 10
}

__global__ void k_write_energy(const float* __restrict__ P, const float* __restrict__ es,
                               float* __restrict__ out){
  out[(size_t)NROW * DIM] =
      -(P[0] / P[3]) * es[0] - (P[1] / P[4]) * es[1] - (P[2] / P[5]) * es[2];
}

// ---------------- host launcher ----------------
extern "C" void kernel_launch(void* const* d_in, const int* in_sizes, int n_in,
                              void* d_out, int out_size, void* d_ws, size_t ws_size,
                              hipStream_t stream) {
  (void)n_in; (void)out_size; (void)ws_size;
  const float* X     = (const float*)d_in[0];
  const int*   c2    = (const int*)  d_in[1];
  const int*   u2    = (const int*)  d_in[2];
  const int*   c3    = (const int*)  d_in[3];
  const int*   u3    = (const int*)  d_in[4];
  const int*   v3    = (const int*)  d_in[5];
  const int*   tt    = (const int*)  d_in[6];
  const float* step  = (const float*)d_in[7];
  const float* gamma = (const float*)d_in[8];
  const float* beta  = (const float*)d_in[9];
  const float* WQ2   = (const float*)d_in[10];
  const float* WK2   = (const float*)d_in[11];
  const float* WQ3   = (const float*)d_in[12];
  const float* WK3   = (const float*)d_in[13];
  const float* Ttau  = (const float*)d_in[14];
  const float* WQm   = (const float*)d_in[15];
  const float* WKm   = (const float*)d_in[16];
  const float* Bmem  = (const float*)d_in[17];
  const float* l2    = (const float*)d_in[18];
  const float* l3    = (const float*)d_in[19];
  const float* lm    = (const float*)d_in[20];
  const float* pb2   = (const float*)d_in[21];
  const float* pb3   = (const float*)d_in[22];
  const float* pbm   = (const float*)d_in[23];
  const int ne = in_sizes[1];
  const int nt = in_sizes[3];
  float* out = (float*)d_out;

  // workspace layout (256B aligned slices)
  char* base = (char*)d_ws;
  size_t off = 0;
  auto take = [&](size_t bytes) -> char* {
    char* p = base + off;
    off = (off + bytes + 255) & ~(size_t)255;
    return p;
  };
  float* params = (float*)take(64 * 4);
  float* esum   = (float*)take(16 * 4);
  float* murstd = (float*)take((size_t)NROW * 2 * 4);
  u32*   m2u    = (u32*)  take((size_t)NROW * 4);
  float* e2buf  = (float*)take((size_t)NROW * 4);
  float* lse2   = (float*)take((size_t)NROW * 4);
  u32*   m3u    = (u32*)  take((size_t)NROW * 4);
  float* e3buf  = (float*)take((size_t)NROW * 4);
  float* lse3   = (float*)take((size_t)NROW * 4);
  float* s3     = (float*)take((size_t)nt * 4);
  float* Km     = (float*)take((size_t)32 * 256 * 4);
  float* s2     = (float*)take((size_t)ne * 4);
  u16*   Wbf    = (u16*)  take((size_t)DIM * NC * 2);
  float* WT     = (float*)take((size_t)NC * DIM * 4);
  u16*   Gbf    = (u16*)  take((size_t)NROW * DIM * 2);
  u16*   Ybf    = (u16*)  take((size_t)NROW * NC * 2);
  float* dY     = (float*)take((size_t)NROW * NC * 4);
  float* dG     = (float*)take((size_t)NROW * DIM * 4);
  // total ~392 MB

  // zero accumulators (graph-capture-safe memsets); m*u 0-init == -max via f2ord
  (void)hipMemsetAsync(esum,  0, 64, stream);
  (void)hipMemsetAsync(m2u,   0, (size_t)NROW * 4, stream);
  (void)hipMemsetAsync(e2buf, 0, (size_t)NROW * 4, stream);
  (void)hipMemsetAsync(m3u,   0, (size_t)NROW * 4, stream);
  (void)hipMemsetAsync(e3buf, 0, (size_t)NROW * 4, stream);
  (void)hipMemsetAsync(dY,    0, (size_t)NROW * NC * 4, stream);

  k_params<<<1, 1, 0, stream>>>(step, l2, l3, lm, pb2, pb3, pbm, params);
  k_prep_w<<<(DIM * NC + 255) / 256, 256, 0, stream>>>(WQ2, WK2, WQ3, WK3, WQm, Wbf, WT);
  k_prep_km<<<32, 256, 0, stream>>>(Bmem, WKm, Km);
  k_ln<<<NROW, 256, 0, stream>>>(X, gamma, beta, Gbf, murstd);

  // forward fused GEMM (bf16 WMMA, 128x64 block tile)
  k_gemm_bf16<<<dim3(NROW / 128, NC / 64), 256, 0, stream>>>(Gbf, Wbf, Ybf);

  // order-2 energy
  k_edge_dot<<<(ne + 7) / 8, 256, 0, stream>>>(Ybf, c2, u2, params, s2, m2u, ne);
  k_seg_sum<<<(ne + 255) / 256, 256, 0, stream>>>(s2, c2, m2u, e2buf, ne);
  k_seg_lse<<<NROW / 256, 256, 0, stream>>>(m2u, e2buf, lse2, esum + 0);

  // order-3 energy
  k_tri_dot<<<(nt + 7) / 8, 256, 0, stream>>>(Ybf, c3, u3, v3, tt, Ttau, params, s3, m3u, nt);
  k_seg_sum<<<(nt + 255) / 256, 256, 0, stream>>>(s3, c3, m3u, e3buf, nt);
  k_seg_lse<<<NROW / 256, 256, 0, stream>>>(m3u, e3buf, lse3, esum + 1);

  // memory-slot energy (fwd + bwd fused, writes dQm region of dY)
  k_mem<<<NROW / 8, 256, 0, stream>>>(Ybf, Km, params, dY, esum + 2);

  // backward scatters (L2-resident atomics)
  k_edge_bwd<<<(ne + 7) / 8, 256, 0, stream>>>(Ybf, c2, u2, s2, lse2, params, dY, ne);
  k_tri_bwd<<<(nt + 7) / 8, 256, 0, stream>>>(Ybf, c3, u3, v3, tt, Ttau, s3, lse3, params, dY, nt);

  // backward fused GEMM (native f32 WMMA, 128x128 block tile -> dY read only 2x)
  k_gemm_f32<<<dim3(NROW / 128, DIM / 128), 256, 0, stream>>>(dY, WT, dG);

  // LN backward + clips + state update, then scalar energy
  k_update<<<NROW, 256, 0, stream>>>(X, gamma, murstd, dG, params, out);
  k_write_energy<<<1, 1, 0, stream>>>(params, esum, out);
}
